// WeightOnlyInt8Linear_2731599200972
// MI455X (gfx1250) — matrix operand order repaired and checked by reading, not compiled, not run on hardware
//
#include <hip/hip_runtime.h>

// WeightOnlyInt8Linear for MI455X (gfx1250, wave32, WMMA f16 path).
// GEMM: M=8192 (B*S), N=11008 (O), K=4096, group size 128 scales.
// Pipeline: global->regs (tile t+1) overlapped with WMMA compute (tile t);
// regs -> convert/dequant -> LDS f16 -> ds_load_b128 fragments -> v_wmma.
// Addressing: 32-bit element offsets incremented by BK per tile (GVS-friendly).

typedef __attribute__((ext_vector_type(16))) _Float16 v16h;
typedef __attribute__((ext_vector_type(8)))  _Float16 v8h;
typedef __attribute__((ext_vector_type(4)))  _Float16 v4h;
typedef __attribute__((ext_vector_type(8)))  float    v8f;

#define GM    8192      // M = B*S
#define GK    4096      // K
#define GO    11008     // N = O
#define NGRP  32        // K / 128 scale groups
#define BM    128
#define BN    128
#define BK    64
#define LDP   72        // LDS row stride in halves (64 + 8 pad; 144B rows, 16B aligned)

__global__ __launch_bounds__(256)
void woq_int8_linear_wmma(const float* __restrict__ x,
                          const int*   __restrict__ w,
                          const float* __restrict__ scales,
                          float*       __restrict__ out) {
    __shared__ __attribute__((aligned(16))) _Float16 As[BM * LDP]; // x tile (m,k) f16
    __shared__ __attribute__((aligned(16))) _Float16 Bs[BN * LDP]; // dequant w tile (n,k) f16

    const int tid  = threadIdx.x;
    const int lane = tid & 31;
    const int wave = tid >> 5;
    const int wm   = wave & 1;   // 0..1 -> 64-row half of the M tile
    const int wn   = wave >> 1;  // 0..3 -> 32-col slice of the N tile

    const int m0 = blockIdx.y * BM;
    const int n0 = blockIdx.x * BN;

    const int lhalf = lane >> 4;        // 0: lanes 0-15, 1: lanes 16-31
    const int lmod  = lane & 15;

    // Per-thread loader coordinates (8 float4/int4 per tile over 128x64 tile),
    // strength-reduced to 32-bit element offsets advanced by BK per k-tile.
    int xoff[8], woff[8], soff[8], loff[8];
    #pragma unroll
    for (int j = 0; j < 8; ++j) {
        const int idx = (j * 256 + tid) * 4;     // 0..8188
        const int row = idx >> 6;
        const int col = idx & 63;
        xoff[j] = (m0 + row) * GK + col;
        woff[j] = (n0 + row) * GK + col;
        soff[j] = (n0 + row) * NGRP;
        loff[j] = row * LDP + col;
    }

    v8f acc[4][2] = {};                 // 4 (M) x 2 (N) tiles of 16x16 f32

    // Pipeline registers for one k-tile of global data.
    float4 xr[8];
    int4   wr[8];
    float  scr[8];

    const int KT = GK / BK;             // 64 k-tiles

    // Preload tile 0 (+ scale group 0).
    #pragma unroll
    for (int j = 0; j < 8; ++j) {
        xr[j] = *(const float4*)(x + xoff[j]); xoff[j] += BK;
        wr[j] = *(const int4*)(w + woff[j]);   woff[j] += BK;
        scr[j] = scales[soff[j]];              soff[j] += 1;
    }

    for (int kt = 0; kt < KT; ++kt) {
        // ---- stage current registers into LDS (convert + dequant) ----
        #pragma unroll
        for (int j = 0; j < 8; ++j) {
            v4h hx;
            hx.x = (_Float16)xr[j].x; hx.y = (_Float16)xr[j].y;
            hx.z = (_Float16)xr[j].z; hx.w = (_Float16)xr[j].w;
            *(v4h*)(&As[loff[j]]) = hx;

            const float sc = scr[j];
            v4h hw;
            hw.x = (_Float16)(sc * (float)wr[j].x);
            hw.y = (_Float16)(sc * (float)wr[j].y);
            hw.z = (_Float16)(sc * (float)wr[j].z);
            hw.w = (_Float16)(sc * (float)wr[j].w);
            *(v4h*)(&Bs[loff[j]]) = hw;
        }
        __syncthreads();

        // ---- issue global loads for tile kt+1; latency hides under WMMAs ----
        if (kt + 1 < KT) {
            #pragma unroll
            for (int j = 0; j < 8; ++j) {
                xr[j] = *(const float4*)(x + xoff[j]); xoff[j] += BK;
                wr[j] = *(const int4*)(w + woff[j]);   woff[j] += BK;
            }
            // New 128-wide quant group starts on even (kt+1): reload scales.
            if (((kt + 1) & 1) == 0) {
                #pragma unroll
                for (int j = 0; j < 8; ++j) {
                    scr[j] = scales[soff[j]];
                    soff[j] += 1;
                }
            }
        }
        // Warm L2 two tiles ahead (global_prefetch_b8).
        if (kt + 2 < KT) {
            __builtin_prefetch(x + xoff[0] + BK, 0, 1);
            __builtin_prefetch(w + woff[0] + BK, 0, 1);
        }

        // ---- compute: 2 kk-steps of K=32, 8 WMMAs each ----
        #pragma unroll
        for (int kk = 0; kk < 2; ++kk) {
            // A fragment (16x32 f16): lanes 0-15 hold K 0-15,
            // lanes 16-31 hold K 16-31 (contiguous 16 halves per lane).
            v16h afrag[4];
            #pragma unroll
            for (int i = 0; i < 4; ++i) {
                const int arow = wm * 64 + i * 16 + lmod;
                const _Float16* ap = &As[arow * LDP + kk * 32 + lhalf * 16];
                v8h lo = *(const v8h*)ap;
                v8h hi = *(const v8h*)(ap + 8);
                afrag[i] = __builtin_shufflevector(lo, hi,
                            0,1,2,3,4,5,6,7,8,9,10,11,12,13,14,15);
            }
            // B fragment (32x16 f16): col = lane%16; lanes 0-15 hold K 0-15,
            // lanes 16-31 hold K 16-31 (contiguous 16 halves per lane).
            v16h bfrag[2];
            #pragma unroll
            for (int jn = 0; jn < 2; ++jn) {
                const int brow = wn * 32 + jn * 16 + lmod;
                const _Float16* bp = &Bs[brow * LDP + kk * 32 + lhalf * 16];
                v8h lo = *(const v8h*)bp;
                v8h hi = *(const v8h*)(bp + 8);
                bfrag[jn] = __builtin_shufflevector(lo, hi,
                            0,1,2,3,4,5,6,7,8,9,10,11,12,13,14,15);
            }
            #pragma unroll
            for (int i = 0; i < 4; ++i) {
                #pragma unroll
                for (int jn = 0; jn < 2; ++jn) {
                    acc[i][jn] = __builtin_amdgcn_wmma_f32_16x16x32_f16(
                        /*neg_a=*/false, afrag[i],
                        /*neg_b=*/false, bfrag[jn],
                        /*c_mod=*/(short)0, acc[i][jn],
                        /*reuse_a=*/false, /*reuse_b=*/false);
                }
            }
        }
        __syncthreads();
    }

    // ---- epilogue: D layout -> row m = r + 8*(lane/16), col n = lane%16 ----
    #pragma unroll
    for (int i = 0; i < 4; ++i) {
        #pragma unroll
        for (int jn = 0; jn < 2; ++jn) {
            const int mbase = m0 + wm * 64 + i * 16 + lhalf * 8;
            const int ncol  = n0 + wn * 32 + jn * 16 + lmod;
            #pragma unroll
            for (int r = 0; r < 8; ++r) {
                out[(mbase + r) * GO + ncol] = acc[i][jn][r];
            }
        }
    }
}

extern "C" void kernel_launch(void* const* d_in, const int* in_sizes, int n_in,
                              void* d_out, int out_size, void* d_ws, size_t ws_size,
                              hipStream_t stream) {
    const float* x      = (const float*)d_in[0];
    const int*   w      = (const int*)d_in[1];
    const float* scales = (const float*)d_in[2];
    float*       out    = (float*)d_out;

    dim3 grid(GO / BN, GM / BM);   // (86, 64)
    dim3 block(256);               // 8 waves (wave32)
    woq_int8_linear_wmma<<<grid, block, 0, stream>>>(x, w, scales, out);
}
